// relScaleDotProductAttention_20220706030069
// MI455X (gfx1250) — matrix operand order repaired and checked by reading, not compiled, not run on hardware
//
#include <hip/hip_runtime.h>
#include <hip/hip_bf16.h>

typedef __attribute__((ext_vector_type(16))) __bf16 v16bf;
typedef __attribute__((ext_vector_type(8)))  __bf16 v8bf;
typedef __attribute__((ext_vector_type(4)))  __bf16 v4bf;
typedef __attribute__((ext_vector_type(8)))  float  v8f;

namespace {
constexpr int kBH = 16;
constexpr int kN  = 2048;
constexpr int kE  = 128;
constexpr int kTileN       = 64;                    // keys per iteration
constexpr int kWaves       = 4;                     // waves per workgroup
constexpr int kRowsPerWave = 16;
constexpr int kRowsPerWG   = kWaves * kRowsPerWave; // 64
constexpr int kKtStride = 136;  // K tile row stride (bf16 elems): 272B, 16B-aligned, distinct banks
constexpr int kPStride  = 72;   // P tile row stride (bf16 elems): 144B, 16B-aligned, distinct banks
constexpr int kMStride  = 68;   // mask tile row stride (f32): 272B -> halves 32 banks apart
}

static __device__ __forceinline__ __bf16 tobf(float f) { return (__bf16)f; }

// Build a v16bf from two 16-byte LDS chunks (each 8 bf16).
static __device__ __forceinline__ v16bf ld2(const __bf16* p0, const __bf16* p1) {
  v8bf a = *(const v8bf*)p0;
  v8bf b = *(const v8bf*)p1;
  return __builtin_shufflevector(a, b, 0,1,2,3,4,5,6,7,8,9,10,11,12,13,14,15);
}

// Swizzled index into sV (bf16 [128 feats][64 keys], 8-key chunks of 16B).
// sw = chunk ^ ((f>>1)&7): for B loads f = e*16+ln this equals chunk^((ln>>1)&7)
// -> independent of e (bases + e*2048B immediates), 16 lanes on 16 distinct
// bank-quads (conflict-free b128).
static __device__ __forceinline__ int vidx(int f, int chunk, int within) {
  const int sw = chunk ^ ((f >> 1) & 7);
  return f * 64 + sw * 8 + within;
}

// 16-lane reductions entirely in VALU via DPP16 (each stage stays inside a
// 16-lane half): quad xor1, quad xor2, half-mirror, mirror.
template <int CTRL>
static __device__ __forceinline__ float dppmov(float x) {
  return __builtin_bit_cast(float,
      __builtin_amdgcn_update_dpp(0, __builtin_bit_cast(int, x),
                                  CTRL, 0xF, 0xF, true));
}
static __device__ __forceinline__ float redmax16(float t) {
  t = fmaxf(t, dppmov<0xB1>(t));
  t = fmaxf(t, dppmov<0x4E>(t));
  t = fmaxf(t, dppmov<0x141>(t));
  t = fmaxf(t, dppmov<0x140>(t));
  return t;
}
static __device__ __forceinline__ float redsum16(float t) {
  t += dppmov<0xB1>(t);
  t += dppmov<0x4E>(t);
  t += dppmov<0x141>(t);
  t += dppmov<0x140>(t);
  return t;
}

__global__ __launch_bounds__(128) void
fa_fwd_bf16_wmma(const float* __restrict__ q, const float* __restrict__ kmat,
                 const float* __restrict__ v, const float* __restrict__ mask,
                 float* __restrict__ out)
{
  __shared__ __bf16 sK[kTileN * kKtStride];    // bf16 K tile, row-major padded
  __shared__ __bf16 sV[kE * kTileN];           // bf16 V tile, transposed + swizzled
  __shared__ __bf16 sP[kWaves][16 * kPStride]; // per-wave P staging
  __shared__ float  sM[kWaves][16 * kMStride]; // per-wave mask tile (async dest)

  const int tid  = threadIdx.x;
  const int wv   = tid >> 5;
  const int lane = tid & 31;
  const int ln   = lane & 15;
  const int h    = lane >> 4;
  const int t4   = tid & 31;

  const int bh   = blockIdx.y;
  const int row0 = blockIdx.x * kRowsPerWG + wv * kRowsPerWave;

  const float* qb = q    + (size_t)bh * kN * kE;
  const float* kb = kmat + (size_t)bh * kN * kE;
  const float* vb = v    + (size_t)bh * kN * kE;
  const float* mb = mask + (size_t)bh * kN * kN;
  float*       ob = out  + (size_t)bh * kN * kE;

  constexpr float scale = 0.08838834764831845f; // 1/sqrt(128)

  // ---- Load 16x128 Q strip, pre-scale, pack into 4 bf16 A-operands.
  v16bf aq[4];
  {
    const float4* q4 = (const float4*)(qb + (size_t)(row0 + ln) * kE);
    #pragma unroll
    for (int kc = 0; kc < 4; ++kc) {
      const int c0 = kc * 8 + h * 2;
      float4 f0 = q4[c0],     f1 = q4[c0 + 1];
      float4 f2 = q4[c0 + 4], f3 = q4[c0 + 5];
      float t[16] = {f0.x,f0.y,f0.z,f0.w, f1.x,f1.y,f1.z,f1.w,
                     f2.x,f2.y,f2.z,f2.w, f3.x,f3.y,f3.z,f3.w};
      #pragma unroll
      for (int j = 0; j < 16; ++j) aq[kc][j] = tobf(t[j] * scale);
    }
  }

  v8f o[8] = {};
  float m[8], l[8];
  #pragma unroll
  for (int r = 0; r < 8; ++r) { m[r] = -__builtin_inff(); l[r] = 0.0f; }

  for (int n0 = 0; n0 < kN; n0 += kTileN) {
    __syncthreads();
    // ---- Stage K: wave-uniform key row per step; b64 stores, conflict-free.
    {
      const float4* k4 = (const float4*)(kb + (size_t)n0 * kE);
      #pragma unroll 4
      for (int i = 0; i < 16; ++i) {
        const int key = 4 * i + wv;
        float4 kf = k4[key * 32 + t4];
        v4bf kp = { tobf(kf.x), tobf(kf.y), tobf(kf.z), tobf(kf.w) };
        *(v4bf*)&sK[key * kKtStride + t4 * 4] = kp;
      }
      // ---- Stage V: 4x4 register transpose into swizzled layout, b64 stores.
      const float4* v4 = (const float4*)(vb + (size_t)n0 * kE);
      for (int g = 0; g < 4; ++g) {
        const int kb4 = wv * 16 + g * 4;
        float4 a0 = v4[(kb4 + 0) * 32 + t4];
        float4 a1 = v4[(kb4 + 1) * 32 + t4];
        float4 a2 = v4[(kb4 + 2) * 32 + t4];
        float4 a3 = v4[(kb4 + 3) * 32 + t4];
        const int f0 = t4 * 4;
        v4bf p0 = { tobf(a0.x), tobf(a1.x), tobf(a2.x), tobf(a3.x) };
        v4bf p1 = { tobf(a0.y), tobf(a1.y), tobf(a2.y), tobf(a3.y) };
        v4bf p2 = { tobf(a0.z), tobf(a1.z), tobf(a2.z), tobf(a3.z) };
        v4bf p3 = { tobf(a0.w), tobf(a1.w), tobf(a2.w), tobf(a3.w) };
        *(v4bf*)&sV[vidx(f0 + 0, kb4 >> 3, kb4 & 7)] = p0;
        *(v4bf*)&sV[vidx(f0 + 1, kb4 >> 3, kb4 & 7)] = p1;
        *(v4bf*)&sV[vidx(f0 + 2, kb4 >> 3, kb4 & 7)] = p2;
        *(v4bf*)&sV[vidx(f0 + 3, kb4 >> 3, kb4 & 7)] = p3;
      }
    }
    __syncthreads();

    // ---- Kick off async global->LDS copy of this tile's mask block:
    // 16 rows x 64 cols f32 per wave, 8 lane-wide B128 ops, ASYNCcnt-tracked,
    // zero VGPRs held. Instruction ib covers rows {2*ib+h}, 16B chunk = ln.
    {
      const float* g0 = mb + (size_t)(row0 + h) * kN + n0 + ln * 4;
      const unsigned l0 =
          (unsigned)(uintptr_t)&sM[wv][h * kMStride + ln * 4];
      #pragma unroll
      for (int ib = 0; ib < 8; ++ib) {
        const float* ga = g0 + (size_t)(2 * ib) * kN;
        const unsigned la = l0 + (unsigned)(2 * ib * kMStride * 4);
        asm volatile("global_load_async_to_lds_b128 %0, %1, off th:TH_LOAD_NT"
                     :: "v"(la), "v"(ga) : "memory");
      }
    }
    // ---- Prefetch next tile's mask block into L2 (32 x 128B lines).
    if (n0 + kTileN < kN) {
      __builtin_prefetch(mb + (size_t)(row0 + ln) * kN + (n0 + kTileN) + h * 32,
                         0, 0);
    }

    // ---- GEMM1: S[16x64] = Qs @ K^T. kc-outer / sub-inner so consecutive
    // WMMAs accumulate into different tiles; depth-3 pipelined B loads.
    v8f s[4] = {};
    {
      auto ldK = [&](int i) -> v16bf {
        const int sub = i & 3, kc = i >> 2;
        const __bf16* bp = &sK[(sub * 16 + ln) * kKtStride + kc * 32 + h * 8];
        return ld2(bp, bp + 16);
      };
      v16bf b0 = ldK(0), b1 = ldK(1);
      #pragma unroll
      for (int i = 0; i < 16; ++i) {
        v16bf b2;
        if (i + 2 < 16) b2 = ldK(i + 2);
        s[i & 3] = __builtin_amdgcn_wmma_f32_16x16x32_bf16(false, aq[i >> 2], false, b0,
                                                           (short)0, s[i & 3], false, false);
        b0 = b1; b1 = b2;
      }
    }

    // ---- Wait for async mask copy, add from LDS (conflict-free ds_load_b32).
    asm volatile("s_wait_asynccnt 0x0" ::: "memory");
    {
      const float* mw = &sM[wv][(h * 8) * kMStride + ln];
      #pragma unroll
      for (int r = 0; r < 8; ++r) {
        #pragma unroll
        for (int sub = 0; sub < 4; ++sub)
          s[sub][r] += mw[r * kMStride + sub * 16];
      }
    }

    // ---- Online softmax; row reductions via DPP16 within each lane-half.
    float corr[8];
    #pragma unroll
    for (int r = 0; r < 8; ++r) {
      float t = fmaxf(fmaxf(s[0][r], s[1][r]), fmaxf(s[2][r], s[3][r]));
      t = redmax16(t);
      const float mn = fmaxf(m[r], t);
      corr[r] = __expf(m[r] - mn);
      m[r] = mn;
      float rs = 0.0f;
      #pragma unroll
      for (int sub = 0; sub < 4; ++sub) {
        const float p = __expf(s[sub][r] - mn);
        s[sub][r] = p;
        rs += p;
      }
      rs = redsum16(rs);
      l[r] = l[r] * corr[r] + rs;
    }
    #pragma unroll
    for (int e = 0; e < 8; ++e) {
      #pragma unroll
      for (int r = 0; r < 8; ++r) o[e][r] *= corr[r];
    }

    // ---- C-layout -> A-layout for P via wave-private LDS (in-order DS).
    __bf16* P = &sP[wv][0];
    #pragma unroll
    for (int r = 0; r < 8; ++r) {
      #pragma unroll
      for (int sub = 0; sub < 4; ++sub)
        P[(h * 8 + r) * kPStride + sub * 16 + ln] = tobf(s[sub][r]);
    }
    v16bf ap[2];
    #pragma unroll
    for (int c = 0; c < 2; ++c)
      ap[c] = ld2(&P[ln * kPStride + 32 * c + h * 8],
                  &P[ln * kPStride + 32 * c + 16 + h * 8]);

    // ---- GEMM2: O[16x128] += P[16x64] @ V[64x128]. c-outer / e-inner so
    // consecutive WMMAs hit different o[e]; depth-3 pipelined B loads.
    {
      auto ldV = [&](int i) -> v16bf {
        const int e = i & 7, c = i >> 3;
        const int f  = e * 16 + ln;
        const int q0 = 4 * c + h;
        return ld2(&sV[vidx(f, q0, 0)], &sV[vidx(f, q0 + 2, 0)]);
      };
      v16bf b0 = ldV(0), b1 = ldV(1);
      #pragma unroll
      for (int i = 0; i < 16; ++i) {
        v16bf b2;
        if (i + 2 < 16) b2 = ldV(i + 2);
        o[i & 7] = __builtin_amdgcn_wmma_f32_16x16x32_bf16(false, ap[i >> 3], false, b0,
                                                           (short)0, o[i & 7], false, false);
        b0 = b1; b1 = b2;
      }
    }
  }

  // ---- Epilogue: normalize by l, store (write-once -> non-temporal).
  #pragma unroll
  for (int r = 0; r < 8; ++r) l[r] = 1.0f / l[r];
  #pragma unroll
  for (int e = 0; e < 8; ++e) {
    #pragma unroll
    for (int r = 0; r < 8; ++r) {
      __builtin_nontemporal_store(
          o[e][r] * l[r],
          ob + (size_t)(row0 + h * 8 + r) * kE + e * 16 + ln);
    }
  }
}

extern "C" void kernel_launch(void* const* d_in, const int* in_sizes, int n_in,
                              void* d_out, int out_size, void* d_ws, size_t ws_size,
                              hipStream_t stream) {
  const float* q    = (const float*)d_in[0];
  const float* k    = (const float*)d_in[1];
  const float* v    = (const float*)d_in[2];
  const float* mask = (const float*)d_in[3];
  float* out = (float*)d_out;
  (void)in_sizes; (void)n_in; (void)out_size; (void)d_ws; (void)ws_size;

  dim3 grid(kN / kRowsPerWG, kBH);   // (32, 16)
  dim3 block(kWaves * 32);           // 128 threads = 4 waves
  fa_fwd_bf16_wmma<<<grid, block, 0, stream>>>(q, k, v, mask, out);
}
